// GeometricStructureEmbedding_11622181503286
// MI455X (gfx1250) — matrix operand order repaired and checked by reading, not compiled, not run on hardware
//
#include <hip/hip_runtime.h>
#include <hip/hip_bf16.h>

typedef __attribute__((ext_vector_type(16))) _Float16 v16h;
typedef __attribute__((ext_vector_type(8)))  _Float16 v8h;
typedef __attribute__((ext_vector_type(2)))  _Float16 v2h;
typedef __attribute__((ext_vector_type(8)))  float    v8f;

#define NPTS 384
#define HID  128
#define RS   136           // padded LDS row stride in halves (bank-conflict-free)
#define NTILES ((NPTS/16)*NPTS)   // 9216
#define FACTOR_A 3.8197186342054885f   // 180/(15*pi)
#define EMB_COEF (-0.14391156831212787f) // -2*ln(10000)/128

// ---------------- Stage 1: 3-NN reference vectors ----------------
__global__ void __launch_bounds__(NPTS)
knn_refv_kernel(const float* __restrict__ pts, float* __restrict__ refv) {
  __shared__ float sp[NPTS * 3];
  const int t = threadIdx.x;
  for (int i = t; i < NPTS * 3; i += NPTS) sp[i] = pts[i];
  __syncthreads();

  const float px = sp[t*3+0], py = sp[t*3+1], pz = sp[t*3+2];
  const float sqn = px*px + py*py + pz*pz;

  float bd[4]; int bi[4];
  #pragma unroll
  for (int j = 0; j < 4; ++j) { bd[j] = 3.0e38f; bi[j] = 0; }

  for (int m = 0; m < NPTS; ++m) {
    const float qx = sp[m*3+0], qy = sp[m*3+1], qz = sp[m*3+2];
    const float sqm = qx*qx + qy*qy + qz*qz;
    float d2 = sqn + sqm - 2.0f*(px*qx + py*qy + pz*qz);
    d2 = fmaxf(d2, 0.0f);
    // stable insertion (strict <): equal keys keep lower index, like lax.top_k
    if (d2 < bd[3]) {
      int j = 3;
      while (j > 0 && d2 < bd[j-1]) { bd[j] = bd[j-1]; bi[j] = bi[j-1]; --j; }
      bd[j] = d2; bi[j] = m;
    }
  }
  // entries 1..3 are the 3 nearest (entry 0 is self)
  #pragma unroll
  for (int k = 0; k < 3; ++k) {
    const int idx = bi[k+1];
    refv[t*9 + k*3 + 0] = sp[idx*3+0] - px;
    refv[t*9 + k*3 + 1] = sp[idx*3+1] - py;
    refv[t*9 + k*3 + 2] = sp[idx*3+2] - pz;
  }
}

// ---------------- Stage 2: fused embedding + WMMA projections ----------------
__device__ __forceinline__ v16h cat8(v8h lo, v8h hi) {
  return __builtin_shufflevector(lo, hi, 0,1,2,3,4,5,6,7,8,9,10,11,12,13,14,15);
}
__device__ __forceinline__ v16h loadA(const _Float16* p) {
  return cat8(*reinterpret_cast<const v8h*>(p),
              *reinterpret_cast<const v8h*>(p + 8));
}

__global__ void __launch_bounds__(256)
geom_emb_kernel(const float* __restrict__ pts,
                const float* __restrict__ wd, const float* __restrict__ bdv,
                const float* __restrict__ wa, const float* __restrict__ bav,
                const float* __restrict__ refv,
                float* __restrict__ out) {
  __shared__ __align__(16) _Float16 emb[4 * 16 * RS];

  const int t    = threadIdx.x;
  const int lane = t & 31;
  const int wave = t >> 5;
  const int lw   = lane & 15;     // N column within tile / M row for A frags
  const int hi   = lane >> 4;     // lane-half selector
  const int cb   = wave * 16;     // output-column base for this wave

  // ---- Preload B fragments (B 32x16, contiguous k order: lane=N, element h -> K=16*hi+h)
  v16h Bd[4], Ba[4];
  {
    const float* rd = wd + (size_t)(cb + lw) * HID;
    const float* ra = wa + (size_t)(cb + lw) * HID;
    #pragma unroll
    for (int kk = 0; kk < 4; ++kk) {
      const int c0 = 32*kk + 16*hi;
      v16h fd, fa;
      #pragma unroll
      for (int h = 0; h < 16; ++h) {
        fd[h] = (_Float16)rd[c0 + h];
        fa[h] = (_Float16)ra[c0 + h];
      }
      Bd[kk] = fd; Ba[kk] = fa;
    }
  }
  const float biasSum = bdv[cb + lw] + bav[cb + lw];

  // embedding-compute mapping: 4 matrices x 16 rows x 4 freq-blocks
  const int g = t >> 6;          // 0 = d-emb, 1..3 = a-emb[k]
  const int r = (t >> 2) & 15;   // row within tile
  const int q = t & 3;           // frequency block (16 freqs each)

  for (int T = blockIdx.x; T < NTILES; T += gridDim.x) {
    const int n  = T / (NPTS/16);
    const int m0 = (T % (NPTS/16)) * 16;

    // ---- compute scalar index and 16 sin/cos freq pairs into LDS
    {
      const int m = m0 + r;
      const float pnx = pts[n*3+0], pny = pts[n*3+1], pnz = pts[n*3+2];
      const float pmx = pts[m*3+0], pmy = pts[m*3+1], pmz = pts[m*3+2];
      const float ax = pmx - pnx, ay = pmy - pny, az = pmz - pnz;
      float x;
      if (g == 0) {
        const float sqn = pnx*pnx + pny*pny + pnz*pnz;
        const float sqm = pmx*pmx + pmy*pmy + pmz*pmz;
        float d2 = sqn + sqm - 2.0f*(pnx*pmx + pny*pmy + pnz*pmz);
        x = sqrtf(fmaxf(d2, 0.0f)) * 5.0f;   // / SIGMA_D (0.2)
      } else {
        const int k = g - 1;
        const float rx = refv[n*9 + k*3 + 0];
        const float ry = refv[n*9 + k*3 + 1];
        const float rz = refv[n*9 + k*3 + 2];
        const float cx = ry*az - rz*ay;
        const float cy = rz*ax - rx*az;
        const float cz = rx*ay - ry*ax;
        const float sv = sqrtf(cx*cx + cy*cy + cz*cz);
        const float cv = rx*ax + ry*ay + rz*az;
        x = atan2f(sv, cv) * FACTOR_A;
      }
      _Float16* dst = emb + (g*16 + r) * RS;
      #pragma unroll
      for (int f = 0; f < 16; ++f) {
        const int i = q*16 + f;
        const float dv = __expf(EMB_COEF * (float)i);
        const float om = x * dv;
        float s, c;
        __sincosf(om, &s, &c);
        v2h pk; pk.x = (_Float16)s; pk.y = (_Float16)c;
        *reinterpret_cast<v2h*>(dst + 2*i) = pk;
      }
    }
    __syncthreads();

    // ---- 4 GEMMs: A frags from LDS (contiguous k order K=16*hi+s, same as B), 16 WMMAs.
    // Load the four A fragments of a k-step into DISTINCT live variables so
    // the backend can batch the 8 ds_load_b128s, wait once, and keep WMMAs
    // overlapped with the next k-step's LDS loads.
    v8f accd = {0.f,0.f,0.f,0.f,0.f,0.f,0.f,0.f};
    v8f acc0 = accd, acc1 = accd, acc2 = accd;
    #pragma unroll
    for (int kk = 0; kk < 4; ++kk) {
      const _Float16* base = emb + lw*RS + 32*kk + 16*hi;
      const v16h Adf = loadA(base);
      const v16h A0f = loadA(base + 16*RS);
      const v16h A1f = loadA(base + 32*RS);
      const v16h A2f = loadA(base + 48*RS);
      accd = __builtin_amdgcn_wmma_f32_16x16x32_f16(false, Adf, false, Bd[kk], (short)0, accd, false, false);
      acc0 = __builtin_amdgcn_wmma_f32_16x16x32_f16(false, A0f, false, Ba[kk], (short)0, acc0, false, false);
      acc1 = __builtin_amdgcn_wmma_f32_16x16x32_f16(false, A1f, false, Ba[kk], (short)0, acc1, false, false);
      acc2 = __builtin_amdgcn_wmma_f32_16x16x32_f16(false, A2f, false, Ba[kk], (short)0, acc2, false, false);
    }

    // ---- combine (max over k commutes with shared bias), store non-temporally
    // C/D layout: VGPR j, lanes<16 -> M=j, lanes>=16 -> M=j+8; N = lane%16
    float* op = out + ((size_t)(n*NPTS + m0 + 8*hi)) * HID + (cb + lw);
    #pragma unroll
    for (int j = 0; j < 8; ++j) {
      const float amax = fmaxf(acc0[j], fmaxf(acc1[j], acc2[j]));
      __builtin_nontemporal_store(accd[j] + amax + biasSum, op + (size_t)j * HID);
    }
    __syncthreads();
  }
}

extern "C" void kernel_launch(void* const* d_in, const int* in_sizes, int n_in,
                              void* d_out, int out_size, void* d_ws, size_t ws_size,
                              hipStream_t stream) {
  const float* pts = (const float*)d_in[0];
  const float* wd  = (const float*)d_in[1];
  const float* bd  = (const float*)d_in[2];
  const float* wa  = (const float*)d_in[3];
  const float* ba  = (const float*)d_in[4];
  float* refv = (float*)d_ws;               // 384*9 floats
  float* out  = (float*)d_out;              // 384*384*128 floats

  knn_refv_kernel<<<1, NPTS, 0, stream>>>(pts, refv);
  geom_emb_kernel<<<1152, 256, 0, stream>>>(pts, wd, bd, wa, ba, refv, out);
}
